// GAE_90589450207433
// MI455X (gfx1250) — matrix operand order repaired and checked by reading, not compiled, not run on hardware
//
#include <hip/hip_runtime.h>
#include <hip/hip_bf16.h>

typedef __attribute__((ext_vector_type(16))) __bf16 v16bf;
typedef __attribute__((ext_vector_type(8)))  float  v8f;

#define ATOMIC_ADD_F32(p, v) \
    __hip_atomic_fetch_add((p), (v), __ATOMIC_RELAXED, __HIP_MEMORY_SCOPE_AGENT)

// ---------------------------------------------------------------- utilities
__global__ void k_fill1(float* __restrict__ p, int n) {
    int i = blockIdx.x * blockDim.x + threadIdx.x;
    if (i < n) p[i] = 1.0f;               // deg starts at 1 (self-loop)
}

__global__ void k_degree(const long long* __restrict__ dst,
                         float* __restrict__ deg, long long E) {
    long long i = (long long)blockIdx.x * blockDim.x + threadIdx.x;
    if (i < E) ATOMIC_ADD_F32(&deg[dst[i]], 1.0f);
}

__global__ void k_rsqrt_inplace(float* __restrict__ d, int n) {
    int i = blockIdx.x * blockDim.x + threadIdx.x;
    if (i < n) {
        float v = d[i];
        d[i] = (v > 0.0f) ? rsqrtf(v) : 0.0f;
    }
}

__global__ void k_cvt_bf16(const float* __restrict__ in,
                           __bf16* __restrict__ out, long long n) {
    long long i = (long long)blockIdx.x * blockDim.x + threadIdx.x;
    if (i < n) out[i] = (__bf16)in[i];
}

__global__ void k_relu_bf16(const float* __restrict__ in,
                            __bf16* __restrict__ out, long long n) {
    long long i = (long long)blockIdx.x * blockDim.x + threadIdx.x;
    if (i < n) out[i] = (__bf16)fmaxf(in[i], 0.0f);
}

// out[i,c] = bias[c] + h[i,c] * dinv[i]^2   (self-loop term; also inits output)
// F == 1<<shift (power of two: 256 or 128)
__global__ void k_selfloop_bias(const float* __restrict__ h,
                                const float* __restrict__ dinv,
                                const float* __restrict__ bias,
                                float* __restrict__ out, long long total,
                                int shift) {
    long long idx = (long long)blockIdx.x * blockDim.x + threadIdx.x;
    if (idx < total) {
        int i = (int)(idx >> shift);
        int c = (int)(idx & ((1 << shift) - 1));
        float di = dinv[i];
        out[idx] = bias[c] + h[idx] * di * di;
    }
}

// -------------------------------------------------------------- WMMA GEMM
// C[N,FOUT] = A[N,K](bf16) x Bw[K,FOUT](bf16), fp32 accumulate.
// One wave computes one 16x16 tile. B panel (K x 16 cols) held in registers
// for the wave's whole lifetime; per row tile, ALL K/32 A-tiles are loaded
// into distinct registers first (one load clause, partial waitcnts), then
// K/32 back-to-back WMMAs accumulate.

template <int K>
__device__ __forceinline__ v16bf load_a_tile(const __bf16* __restrict__ ap,
                                             int ks, int hi8) {
    // A 16x32 bf16 layout: elem j -> K = ks*32 + 16*(j>>3) + (j&7) + 8*(lane>>4)
    v16bf a;
#pragma unroll
    for (int i = 0; i < 8; ++i) {
        const int jj = 2 * i;
        const int k  = ks * 32 + ((jj >> 3) << 4) + (jj & 7) + hi8;
        a[jj]     = ap[k];
        a[jj + 1] = ap[k + 1];
    }
    return a;
}

template <int K, int FOUT>
__global__ void k_gemm_bf16(const __bf16* __restrict__ A,
                            const __bf16* __restrict__ Bw,
                            float* __restrict__ C, int N) {
    constexpr int KS = K / 32;
    const int lane    = threadIdx.x & 31;
    const int wave    = threadIdx.x >> 5;
    const int colBase = blockIdx.x * 16;
    const int nlo     = lane & 15;
    const int hiHalf  = lane >> 4;        // 0: lanes 0-15, 1: lanes 16-31
    const int hi8     = hiHalf * 8;
    const int rowTiles = (N + 15) >> 4;

    // Preload B panel: B is 32x16 per step; lane holds column nlo,
    // element j -> K = ks*32 + 16*(j>>3) + (j&7) + 8*hiHalf.
    v16bf breg[KS];
#pragma unroll
    for (int ks = 0; ks < KS; ++ks) {
        v16bf b;
        const int kb = ks * 32 + hiHalf * 8;
#pragma unroll
        for (int j = 0; j < 16; ++j)
            b[j] = Bw[(size_t)(kb + (j & 7) + ((j >> 3) << 4)) * FOUT + colBase + nlo];
        breg[ks] = b;
    }

    for (int rt = blockIdx.y * 4 + wave; rt < rowTiles; rt += gridDim.y * 4) {
        const int  rowBase  = rt << 4;
        const bool fullTile = (rowBase + 16 <= N);   // wave-uniform

        int arow = rowBase + nlo;
        if (!fullTile && arow >= N) arow = N - 1;    // clamp only on tail tile
        const __bf16* ap = A + (size_t)arow * K;     // K constexpr -> shift

        // Issue the whole A row-tile as one load clause (distinct dests so
        // the scheduler can use partial waitcnts and overlap with WMMA).
        v16bf areg[KS];
#pragma unroll
        for (int ks = 0; ks < KS; ++ks)
            areg[ks] = load_a_tile<K>(ap, ks, hi8);

        v8f c = {};
#pragma unroll
        for (int ks = 0; ks < KS; ++ks)
            c = __builtin_amdgcn_wmma_f32_16x16x32_bf16(
                    false, areg[ks], false, breg[ks], (short)0, c, false, false);

        // D layout: VGPR r -> M = rowBase + r + hi8, N = colBase + nlo
        // FOUT constexpr -> stores use immediate offsets.
        float* cp = C + (size_t)(rowBase + hi8) * FOUT + colBase + nlo;
        if (fullTile) {
#pragma unroll
            for (int r = 0; r < 8; ++r)
                cp[r * FOUT] = c[r];
        } else {
#pragma unroll
            for (int r = 0; r < 8; ++r) {
                const int m = rowBase + r + hi8;
                if (m < N) cp[r * FOUT] = c[r];
            }
        }
    }
}

// ------------------------------------------------------- edge scatter-add
// One wave per edge; lanes stride the channel dim -> coalesced gather and
// coalesced global_atomic_add_f32 scatter. Feature tables are L2-resident
// (102 MB / 51 MB vs 192 MB L2), so this runs at L2, not HBM, rates.
template <int F>
__global__ void k_edge_scatter(const float* __restrict__ h,
                               const long long* __restrict__ src,
                               const long long* __restrict__ dst,
                               const float* __restrict__ dinv,
                               float* __restrict__ out, long long E) {
    const int lane = threadIdx.x & 31;
    const long long wid = (long long)blockIdx.x * (blockDim.x >> 5) +
                          (threadIdx.x >> 5);
    const long long nw = (long long)gridDim.x * (blockDim.x >> 5);
    for (long long e = wid; e < E; e += nw) {
        const long long s = src[e];
        const long long d = dst[e];
        // prefetch next edge's source row (global_prefetch)
        const long long en = e + nw;
        if (en < E) __builtin_prefetch(&h[src[en] * F + lane], 0, 0);

        const float norm = dinv[s] * dinv[d];
        const float* hs = h + s * (long long)F;
        float* od = out + d * (long long)F;
#pragma unroll
        for (int r = 0; r < F / 32; ++r)
            ATOMIC_ADD_F32(&od[lane + 32 * r], hs[lane + 32 * r] * norm);
    }
}

// ------------------------------------------------------------------ launch
extern "C" void kernel_launch(void* const* d_in, const int* in_sizes, int n_in,
                              void* d_out, int out_size, void* d_ws, size_t ws_size,
                              hipStream_t stream) {
    const float*     x  = (const float*)d_in[0];
    const long long* ei = (const long long*)d_in[1];
    const float*     W1 = (const float*)d_in[2];
    const float*     b1 = (const float*)d_in[3];
    const float*     W2 = (const float*)d_in[4];
    const float*     b2 = (const float*)d_in[5];

    const int IN  = 256, HID = 256, OUT = 128;
    const int N = in_sizes[0] / IN;
    const long long E = in_sizes[1] / 2;
    const long long NF1 = (long long)N * HID;
    const long long NF2 = (long long)N * OUT;

    // Workspace carve-out (regions reused across phases)
    char* ws = (char*)d_ws;
    size_t o = 0;
    auto carve = [&](size_t bytes) {
        size_t p = o;
        o = (o + bytes + 255) & ~(size_t)255;
        return p;
    };
    float*  dinv = (float*)(ws + carve((size_t)N * 4));          // deg -> dinv
    __bf16* W1b  = (__bf16*)(ws + carve((size_t)IN * HID * 2));
    __bf16* W2b  = (__bf16*)(ws + carve((size_t)HID * OUT * 2));
    __bf16* xbf  = (__bf16*)(ws + carve((size_t)NF1 * 2));       // xbf -> relu(agg1)
    float*  hbuf = (float*)(ws + carve((size_t)NF1 * 4));        // h1  -> h2
    float*  agg1 = (float*)(ws + carve((size_t)NF1 * 4));
    float*  zout = (float*)d_out;

    const long long* srcI = ei;
    const long long* dstI = ei + E;

    // --- normalization: deg (init 1 for self-loop) -> dinv
    k_fill1<<<(N + 255) / 256, 256, 0, stream>>>(dinv, N);
    k_degree<<<(unsigned)((E + 255) / 256), 256, 0, stream>>>(dstI, dinv, E);
    k_rsqrt_inplace<<<(N + 255) / 256, 256, 0, stream>>>(dinv, N);

    // --- bf16 conversions
    k_cvt_bf16<<<(unsigned)((NF1 + 255) / 256), 256, 0, stream>>>(x, xbf, NF1);
    k_cvt_bf16<<<(IN * HID + 255) / 256, 256, 0, stream>>>(W1, W1b, IN * HID);
    k_cvt_bf16<<<(HID * OUT + 255) / 256, 256, 0, stream>>>(W2, W2b, HID * OUT);

    // --- layer 1: h = x @ W1 (WMMA bf16), agg1 = b1 + self-loop + scatter
    k_gemm_bf16<256, 256><<<dim3(HID / 16, 256), 128, 0, stream>>>(
        xbf, W1b, hbuf, N);
    k_selfloop_bias<<<(unsigned)((NF1 + 255) / 256), 256, 0, stream>>>(
        hbuf, dinv, b1, agg1, NF1, 8);
    {
        unsigned blocks = (unsigned)((E + 7) / 8);
        if (blocks > 131072u) blocks = 131072u;
        k_edge_scatter<256><<<blocks, 256, 0, stream>>>(hbuf, srcI, dstI, dinv,
                                                        agg1, E);
    }

    // --- relu + downcast (reuse xbf region as bf16 hidden activations)
    k_relu_bf16<<<(unsigned)((NF1 + 255) / 256), 256, 0, stream>>>(agg1, xbf, NF1);

    // --- layer 2: h2 = relu(agg1) @ W2 (WMMA), out = b2 + self-loop + scatter
    k_gemm_bf16<256, 128><<<dim3(OUT / 16, 256), 128, 0, stream>>>(
        xbf, W2b, hbuf, N);
    k_selfloop_bias<<<(unsigned)((NF2 + 255) / 256), 256, 0, stream>>>(
        hbuf, dinv, b2, zout, NF2, 7);
    {
        unsigned blocks = (unsigned)((E + 7) / 8);
        if (blocks > 131072u) blocks = 131072u;
        k_edge_scatter<128><<<blocks, 256, 0, stream>>>(hbuf, srcI, dstI, dinv,
                                                        zout, E);
    }
}
